// Transformer_40458591928786
// MI455X (gfx1250) — matrix operand order repaired and checked by reading, not compiled, not run on hardware
//
#include <hip/hip_runtime.h>

// ---------------------------------------------------------------------------
// MI455X (gfx1250, wave32) transformer forward.
//
// Quirk-faithful optimization: enc[0] and dec[0] are dead (the reference
// applies every layer to the same input and keeps only the last), so only
// enc[1] and dec[1] are computed.
//
// GEMMs: v_wmma_f32_16x16x32_bf16, fp32 -> bf16 staged through LDS with
// register-staged software pipelining (global loads for chunk k+1 issued
// before the WMMAs of chunk k, wait folded into the next LDS store).
//  - A staged [m][k]: per-lane fragment = two contiguous 8xbf16 runs
//    -> 2x ds_load_b128.
//  - B staged TRANSPOSED [n][k]: per-lane fragment = two contiguous 8xbf16
//    runs -> 2x ds_load_b128. Transpose staging stores go down-k (contiguous)
//    as b32/b64.
//  - Tile configs: 32x64 (wave = 16x16, N=64 GEMMs) and 64x128
//    (wave = 32x32 = 4 WMMAs/K-step, all big GEMMs).
//
// Residual add exercises the gfx1250 async path:
// global_load_async_to_lds_b128 + s_wait_asynccnt.
//
// Workspace requirement: 18 * 4096*1024 floats = ~302 MB.
// ---------------------------------------------------------------------------

typedef __attribute__((ext_vector_type(16))) __bf16 bf16x16;
typedef __attribute__((ext_vector_type(8)))  __bf16 bf16x8;
typedef __attribute__((ext_vector_type(4)))  __bf16 bf16x4;
typedef __attribute__((ext_vector_type(2)))  __bf16 bf16x2;
typedef __attribute__((ext_vector_type(8)))  float  f32x8;

constexpr int NB  = 4;      // batch
constexpr int NT  = 1024;   // seq len
constexpr int ND  = 1024;   // d_model
constexpr int NH  = 16;     // heads
constexpr int NHS = 64;     // head size
constexpr int NV  = 32000;  // vocab
constexpr int NFF = 5120;   // ffn hidden
constexpr float LN_EPS = 1e-5f;

constexpr long long ROWS = (long long)NB * NT;   // 4096
constexpr long long SZ   = ROWS * ND;            // 4,194,304 floats (16 MB)

// ---------------------------------------------------------------------------
// Batched WMMA GEMM:  C = act( A @ B (+ bias) )
//  A: [M,K] row-major (lda); B: [K,N] row-major (ldb) or, if BTR, [N,K] (ldb)
//  C: [M,N] row-major (ldc).
//  Batch z: off = (z/inner)*outer + (z%inner)*inner per operand.
//  Block = 256 threads = 8 waves as 2(M) x 4(N).
//  Block tile = (32*WMT) x (64*WNT); wave tile = (16*WMT) x (16*WNT).
//  Requires M % (32*WMT) == 0, N % (64*WNT) == 0, K % 32 == 0.
// ---------------------------------------------------------------------------
template<int WMT, int WNT, bool BTR, bool HASB, bool GELU_ACT>
__global__ __launch_bounds__(256) void wmma_gemm_ker(
    const float* __restrict__ Ag, const float* __restrict__ Bg,
    const float* __restrict__ biasg, float* __restrict__ Cg,
    int M, int N, int K, int lda, int ldb, int ldc,
    long long Ao, long long Ai, long long Bo, long long Bi,
    long long Co, long long Ci, long long bo, long long bi,
    int inner)
{
  constexpr int BM = 32 * WMT;
  constexpr int BN = 64 * WNT;
  constexpr int AR = BM / 32;   // A float4 loads per thread per K-chunk
  constexpr int BR = BN / 32;   // B k-rows per thread (transpose staging)

  const int z  = blockIdx.z;
  const int zo = z / inner, zi = z % inner;
  const float* A  = Ag + zo * Ao + zi * Ai;
  const float* Bp = Bg + zo * Bo + zi * Bi;
  float*       C  = Cg + zo * Co + zi * Ci;
  const float* bias = nullptr;
  if (HASB) bias = biasg + zo * bo + zi * bi;

  // rows padded to 40 bf16 (80 B, 16-byte aligned) for wide conflict-lean IO
  __shared__ __attribute__((aligned(16))) __bf16 As[BM][40];  // [m][k]
  __shared__ __attribute__((aligned(16))) __bf16 Bs[BN][40];  // [n][k] (transposed)

  const int tid  = threadIdx.x;
  const int wave = tid >> 5, lane = tid & 31;
  const int wm   = wave >> 2, wn = wave & 3;
  const int hlf  = lane >> 4, l16 = lane & 15;
  const int tileM = blockIdx.y * BM;
  const int tileN = blockIdx.x * BN;

  // staging registers (max sizes; unused entries optimized away)
  float4 aR[2];
  float4 bR[4];       // BTR path
  float  bT[4][4];    // non-BTR transpose path: [k-row][n-col]

  const int cg = tid % (BN / 4), rg = tid / (BN / 4);  // non-BTR mapping
  const int r0 = rg * BR, c0 = cg * 4;

  auto loadA = [&](int k0) {
#pragma unroll
    for (int i = 0; i < AR; ++i) {
      const int idx = tid + i * 256;
      const int r = idx >> 3, c = (idx & 7) * 4;
      aR[i] = *reinterpret_cast<const float4*>(
          A + (long long)(tileM + r) * lda + (k0 + c));
    }
  };
  auto storeA = [&]() {
#pragma unroll
    for (int i = 0; i < AR; ++i) {
      const int idx = tid + i * 256;
      const int r = idx >> 3, c = (idx & 7) * 4;
      const bf16x4 w = { (__bf16)aR[i].x, (__bf16)aR[i].y,
                         (__bf16)aR[i].z, (__bf16)aR[i].w };
      *reinterpret_cast<bf16x4*>(&As[r][c]) = w;
    }
  };
  auto loadB = [&](int k0) {
    if (BTR) {
#pragma unroll
      for (int i = 0; i < BN / 64; ++i) {
        const int idx = tid + i * 256;
        const int n = idx >> 2, kk = (idx & 3) * 8;
        const float* src = Bp + (long long)(tileN + n) * ldb + (k0 + kk);
        bR[2 * i]     = *reinterpret_cast<const float4*>(src);
        bR[2 * i + 1] = *reinterpret_cast<const float4*>(src + 4);
      }
    } else {
#pragma unroll
      for (int j = 0; j < BR; ++j) {
        const float4 v = *reinterpret_cast<const float4*>(
            Bp + (long long)(k0 + r0 + j) * ldb + (tileN + c0));
        bT[j][0] = v.x; bT[j][1] = v.y; bT[j][2] = v.z; bT[j][3] = v.w;
      }
    }
  };
  auto storeB = [&]() {
    if (BTR) {
#pragma unroll
      for (int i = 0; i < BN / 64; ++i) {
        const int idx = tid + i * 256;
        const int n = idx >> 2, kk = (idx & 3) * 8;
        const float4 v0 = bR[2 * i], v1 = bR[2 * i + 1];
        const bf16x8 w = { (__bf16)v0.x, (__bf16)v0.y, (__bf16)v0.z, (__bf16)v0.w,
                           (__bf16)v1.x, (__bf16)v1.y, (__bf16)v1.z, (__bf16)v1.w };
        *reinterpret_cast<bf16x8*>(&Bs[n][kk]) = w;
      }
    } else {
      // store down-k (contiguous axis of Bs): wide b32/b64 stores
#pragma unroll
      for (int j = 0; j < 4; ++j) {
        if (BR == 2) {
          const bf16x2 w = { (__bf16)bT[0][j], (__bf16)bT[1][j] };
          *reinterpret_cast<bf16x2*>(&Bs[c0 + j][r0]) = w;
        } else {
          const bf16x4 w = { (__bf16)bT[0][j], (__bf16)bT[1][j],
                             (__bf16)bT[2][j], (__bf16)bT[3][j] };
          *reinterpret_cast<bf16x4*>(&Bs[c0 + j][r0]) = w;
        }
      }
    }
  };

  const f32x8 zf = {};
  f32x8 acc[WMT][WNT];
#pragma unroll
  for (int i = 0; i < WMT; ++i)
#pragma unroll
    for (int j = 0; j < WNT; ++j) acc[i][j] = zf;

  // ---- software-pipelined main loop ---------------------------------------
  loadA(0); loadB(0);
  for (int k0 = 0; k0 < K; k0 += 32) {
    storeA(); storeB();
    __syncthreads();
    if (k0 + 32 < K) { loadA(k0 + 32); loadB(k0 + 32); }  // overlap with WMMA

    // A lane layout: af[0..7] = k(hlf*8 .. +7), af[8..15] = k(16+hlf*8 .. +7)
    // B lane layout: bfr[0..7] = k(hlf*8 .. +7), bfr[8..15] = k(16+hlf*8 .. +7)
    bf16x16 af[WMT], bfr[WNT];
#pragma unroll
    for (int mi = 0; mi < WMT; ++mi) {
      const int am = wm * (16 * WMT) + mi * 16 + l16;
      const bf16x8 a0 = *reinterpret_cast<const bf16x8*>(&As[am][hlf * 8]);
      const bf16x8 a1 = *reinterpret_cast<const bf16x8*>(&As[am][16 + hlf * 8]);
      af[mi] = __builtin_shufflevector(a0, a1, 0, 1, 2, 3, 4, 5, 6, 7,
                                       8, 9, 10, 11, 12, 13, 14, 15);
    }
#pragma unroll
    for (int ni = 0; ni < WNT; ++ni) {
      const int bn = wn * (16 * WNT) + ni * 16 + l16;
      const bf16x8 b0 = *reinterpret_cast<const bf16x8*>(&Bs[bn][hlf * 8]);
      const bf16x8 b1 = *reinterpret_cast<const bf16x8*>(&Bs[bn][16 + hlf * 8]);
      bfr[ni] = __builtin_shufflevector(b0, b1, 0, 1, 2, 3, 4, 5, 6, 7,
                                        8, 9, 10, 11, 12, 13, 14, 15);
    }
#pragma unroll
    for (int mi = 0; mi < WMT; ++mi)
#pragma unroll
      for (int ni = 0; ni < WNT; ++ni)
        acc[mi][ni] = __builtin_amdgcn_wmma_f32_16x16x32_bf16(
            false, af[mi], false, bfr[ni], (short)0, acc[mi][ni], false, false);
    __syncthreads();
  }

  // ---- epilogue: C/D layout VGPR r -> M = r + 8*hlf, n = lane%16 ----------
#pragma unroll
  for (int mi = 0; mi < WMT; ++mi) {
#pragma unroll
    for (int ni = 0; ni < WNT; ++ni) {
      const int n_out = tileN + wn * (16 * WNT) + ni * 16 + l16;
      float bv = 0.f;
      if (HASB) bv = bias[n_out];
#pragma unroll
      for (int r = 0; r < 8; ++r) {
        const int m_out = tileM + wm * (16 * WMT) + mi * 16 + (hlf << 3) + r;
        float v = acc[mi][ni][r];
        if (HASB) v += bv;
        if (GELU_ACT) v = 0.5f * v * (1.0f + erff(v * 0.70710678118654752f));
        C[(long long)m_out * ldc + n_out] = v;
      }
    }
  }
}

// ---------------------------------------------------------------------------
// Elementwise / row kernels
// ---------------------------------------------------------------------------
__global__ __launch_bounds__(256) void embed_ker(
    const int* __restrict__ idx, const float* __restrict__ tok,
    const float* __restrict__ pos, float* __restrict__ out)
{
  const int row = blockIdx.x;            // b*NT + t
  const int t   = row % NT;
  const long long to = (long long)idx[row] * ND;
  const long long po = (long long)t * ND;
  const long long oo = (long long)row * ND;
#pragma unroll
  for (int j = 0; j < 4; ++j) {
    const int d = threadIdx.x + j * 256;
    out[oo + d] = tok[to + d] + pos[po + d];
  }
}

// Residual add, staged through LDS with the gfx1250 async-copy path:
// global_load_async_to_lds_b128 (ASYNCcnt) + s_wait_asynccnt.
__global__ __launch_bounds__(256) void add_ker(
    const float4* __restrict__ a, const float4* __restrict__ b,
    float4* __restrict__ o)
{
  __shared__ __attribute__((aligned(16))) float4 bufA[256];
  __shared__ __attribute__((aligned(16))) float4 bufB[256];
  const int tid = threadIdx.x;
  const long long i = (long long)blockIdx.x * 256 + tid;

  const unsigned ldsA = (unsigned)(unsigned long long)(uintptr_t)&bufA[tid];
  const unsigned ldsB = (unsigned)(unsigned long long)(uintptr_t)&bufB[tid];
  asm volatile("global_load_async_to_lds_b128 %0, %1, off"
               :: "v"(ldsA), "v"(a + i) : "memory");
  asm volatile("global_load_async_to_lds_b128 %0, %1, off"
               :: "v"(ldsB), "v"(b + i) : "memory");
  asm volatile("s_wait_asynccnt 0" ::: "memory");

  const float4 x = bufA[tid], y = bufB[tid];
  o[i] = make_float4(x.x + y.x, x.y + y.y, x.z + y.z, x.w + y.w);
}

template<bool ADDIN>
__global__ __launch_bounds__(256) void ln_ker(
    const float* __restrict__ in, const float* __restrict__ g,
    const float* __restrict__ b, float* __restrict__ out)
{
  const int row = blockIdx.x;
  const float* p = in + (long long)row * ND;
  float* o = out + (long long)row * ND;
  __shared__ float red[256];
  const int tid = threadIdx.x;
  float x[4]; float s = 0.f;
#pragma unroll
  for (int j = 0; j < 4; ++j) { x[j] = p[tid + j * 256]; s += x[j]; }
  red[tid] = s; __syncthreads();
  for (int off = 128; off > 0; off >>= 1) {
    if (tid < off) red[tid] += red[tid + off];
    __syncthreads();
  }
  const float mean = red[0] * (1.0f / ND);
  __syncthreads();
  float sq = 0.f;
#pragma unroll
  for (int j = 0; j < 4; ++j) { const float d = x[j] - mean; sq += d * d; }
  red[tid] = sq; __syncthreads();
  for (int off = 128; off > 0; off >>= 1) {
    if (tid < off) red[tid] += red[tid + off];
    __syncthreads();
  }
  const float inv = rsqrtf(red[0] * (1.0f / ND) + LN_EPS);
#pragma unroll
  for (int j = 0; j < 4; ++j) {
    const int d = tid + j * 256;
    const float nv = (x[j] - mean) * inv * g[d] + b[d];
    o[d] = ADDIN ? (x[j] + nv) : nv;
  }
}

// scores softmax over s-dim; faithful quirk: scale = sqrt(HS) = 8 (multiply)
__global__ __launch_bounds__(256) void softmax_ker(
    float* __restrict__ S, float scale, int causal)
{
  const int row = blockIdx.x;            // (b*NT + t) for current head slice
  const int t   = row % NT;
  float* p = S + (long long)row * NT;
  __shared__ float red[256];
  const int tid = threadIdx.x;
  float v[4]; float mx = -3.402823466e38f;
#pragma unroll
  for (int j = 0; j < 4; ++j) {
    const int s = tid + j * 256;
    float x = p[s] * scale;
    if (causal && s > t) x = -__builtin_inff();
    v[j] = x; mx = fmaxf(mx, x);
  }
  red[tid] = mx; __syncthreads();
  for (int off = 128; off > 0; off >>= 1) {
    if (tid < off) red[tid] = fmaxf(red[tid], red[tid + off]);
    __syncthreads();
  }
  mx = red[0]; __syncthreads();
  float sum = 0.f;
#pragma unroll
  for (int j = 0; j < 4; ++j) { v[j] = __expf(v[j] - mx); sum += v[j]; }
  red[tid] = sum; __syncthreads();
  for (int off = 128; off > 0; off >>= 1) {
    if (tid < off) red[tid] += red[tid + off];
    __syncthreads();
  }
  const float inv = 1.0f / red[0];
#pragma unroll
  for (int j = 0; j < 4; ++j) p[tid + j * 256] = v[j] * inv;
}

// ---------------------------------------------------------------------------
// Host-side orchestration
// ---------------------------------------------------------------------------
struct AttP  { const float *wq, *bq, *wk, *bk, *wv, *wo, *bo; };
struct LayerP{ AttP att; const float *w1, *b1, *w2, *b2, *g1, *b1n, *g2, *b2n; };

static LayerP load_layer(void* const* d_in, int base) {
  auto F = [&](int i) { return (const float*)d_in[i]; };
  LayerP L;
  L.att.wq = F(base + 0); L.att.bq = F(base + 1);
  L.att.wk = F(base + 2); L.att.bk = F(base + 3);
  L.att.wv = F(base + 4); L.att.wo = F(base + 5); L.att.bo = F(base + 6);
  L.w1 = F(base + 7);  L.b1  = F(base + 8);
  L.w2 = F(base + 9);  L.b2  = F(base + 10);
  L.g1 = F(base + 11); L.b1n = F(base + 12);
  L.g2 = F(base + 13); L.b2n = F(base + 14);
  return L;
}

template<int WMT, int WNT, bool BTR, bool HASB, bool GELU_ACT>
static void rgemm(hipStream_t st, const float* A, const float* Bm,
                  const float* bias, float* C,
                  int M, int N, int K, int lda, int ldb, int ldc,
                  long long Ao = 0, long long Ai = 0,
                  long long Bo = 0, long long Bi = 0,
                  long long Co = 0, long long Ci = 0,
                  long long bo = 0, long long bi = 0,
                  int inner = 1, int nz = 1)
{
  dim3 grid(N / (64 * WNT), M / (32 * WMT), nz);
  wmma_gemm_ker<WMT, WNT, BTR, HASB, GELU_ACT><<<grid, dim3(256), 0, st>>>(
      A, Bm, bias, C, M, N, K, lda, ldb, ldc,
      Ao, Ai, Bo, Bi, Co, Ci, bo, bi, inner);
}

static void run_add(hipStream_t st, const float* a, const float* b, float* o) {
  add_ker<<<(int)(SZ / 4 / 256), 256, 0, st>>>(
      (const float4*)a, (const float4*)b, (float4*)o);
}

// Q/K/V layout: [B,H,T,HS]; attention context Ob: [B,T,D]; outp: [B,T,D]
static void attention(hipStream_t st, const float* xin, const AttP& p,
                      bool causal, float* Qb, float* Kb, float* Vb,
                      float* Sb, float* Ob, float* outp)
{
  const long long TD  = (long long)NT * ND;
  const long long DH  = (long long)ND * NHS;
  const long long HTH = (long long)NH * NT * NHS;
  const long long TH  = (long long)NT * NHS;
  const long long TT  = (long long)NT * NT;

  // per-(b,h) projections: z = b*H + h  (N = 64 -> small tile config)
  rgemm<1, 1, false, true , false>(st, xin, p.wq, p.bq, Qb, NT, NHS, ND, ND, NHS, NHS,
                                   TD, 0, 0, DH, HTH, TH, 0, NHS, NH, NB * NH);
  rgemm<1, 1, false, true , false>(st, xin, p.wk, p.bk, Kb, NT, NHS, ND, ND, NHS, NHS,
                                   TD, 0, 0, DH, HTH, TH, 0, NHS, NH, NB * NH);
  rgemm<1, 1, false, false, false>(st, xin, p.wv, nullptr, Vb, NT, NHS, ND, ND, NHS, NHS,
                                   TD, 0, 0, DH, HTH, TH, 0, 0, NH, NB * NH);

  // one head at a time through a 16 MB score slice (keeps workspace small)
  for (int h = 0; h < NH; ++h) {
    // S = Q @ K^T : K arrives [N=s, K=c] -> BTR (contiguous transpose staging)
    rgemm<2, 2, true , false, false>(st, Qb + (long long)h * TH, Kb + (long long)h * TH,
                                     nullptr, Sb, NT, NT, NHS, NHS, NHS, NT,
                                     HTH, 0, HTH, 0, TT, 0, 0, 0, 1, NB);
    softmax_ker<<<NB * NT, 256, 0, st>>>(Sb, 8.0f /* sqrt(HS) quirk */,
                                         causal ? 1 : 0);
    rgemm<1, 1, false, false, false>(st, Sb, Vb + (long long)h * TH, nullptr,
                                     Ob + (long long)h * NHS, NT, NHS, NT, NT, NHS, ND,
                                     TT, 0, HTH, 0, TD, 0, 0, 0, 1, NB);
  }
  rgemm<2, 2, false, true, false>(st, Ob, p.wo, p.bo, outp,
                                  (int)ROWS, ND, ND, ND, ND, ND);
}

static void ffn(hipStream_t st, const float* xin, const LayerP& L,
                float* FFHb, float* outp)
{
  rgemm<2, 2, false, true, true >(st, xin, L.w1, L.b1, FFHb,
                                  (int)ROWS, NFF, ND, ND, NFF, NFF);
  rgemm<2, 2, false, true, false>(st, FFHb, L.w2, L.b2, outp,
                                  (int)ROWS, ND, NFF, NFF, ND, ND);
}

extern "C" void kernel_launch(void* const* d_in, const int* in_sizes, int n_in,
                              void* d_out, int out_size, void* d_ws,
                              size_t ws_size, hipStream_t stream)
{
  (void)in_sizes; (void)n_in; (void)out_size; (void)ws_size;

  // input flattening: insertion order of the setup_inputs() pytree
  // 0:idx 1:tok 2:pos 3..17:enc[0] 18..32:enc[1] 33..47:dec[0] 48..62:dec[1]
  // 63:gf 64:bf 65:wf 66:bfw
  const int*   idx = (const int*)d_in[0];
  const float* tok = (const float*)d_in[1];
  const float* pos = (const float*)d_in[2];
  const LayerP ENC = load_layer(d_in, 3 + 15);   // enc[1] (only live enc layer)
  const LayerP DEC = load_layer(d_in, 3 + 45);   // dec[1] (only live dec layer)
  const float* gf  = (const float*)d_in[63];
  const float* bff = (const float*)d_in[64];
  const float* wf  = (const float*)d_in[65];
  const float* bfw = (const float*)d_in[66];

  float* ws  = (float*)d_ws;
  float* X    = ws + 0  * SZ;   // embedding
  float* XOUT = ws + 1  * SZ;   // encoder output
  float* Qb   = ws + 2  * SZ;
  float* Kb   = ws + 3  * SZ;
  float* Vb   = ws + 4  * SZ;
  float* Sb   = ws + 5  * SZ;   // per-head score slice (B*T*T == SZ)
  float* Ob   = ws + 6  * SZ;   // attn context
  float* T1   = ws + 7  * SZ;
  float* T2   = ws + 8  * SZ;
  float* T3   = ws + 9  * SZ;
  float* T4   = ws + 10 * SZ;
  float* T5   = ws + 11 * SZ;
  float* T6   = ws + 12 * SZ;
  float* FFH  = ws + 13 * SZ;   // 5*SZ (4096 x 5120)

  // ---- embedding -----------------------------------------------------------
  embed_ker<<<(int)ROWS, 256, 0, stream>>>(idx, tok, pos, X);

  // ---- encoder layer (enc[1]) ---------------------------------------------
  attention(stream, X, ENC.att, /*causal=*/false, Qb, Kb, Vb, Sb, Ob, T1);
  run_add(stream, X, T1, T2);                                 // x + attn
  ln_ker<false><<<(int)ROWS, 256, 0, stream>>>(T2, ENC.g1, ENC.b1n, T3);
  ffn(stream, T3, ENC, FFH, T1);
  run_add(stream, T3, T1, T2);                                // x + ffn
  ln_ker<false><<<(int)ROWS, 256, 0, stream>>>(T2, ENC.g2, ENC.b2n, XOUT);

  // ---- decoder layer (dec[1]); src = X, trg = XOUT ------------------------
  attention(stream, X, DEC.att, /*causal=*/true, Qb, Kb, Vb, Sb, Ob, T1);
  run_add(stream, X, T1, T2);                                 // src + attn
  ln_ker<true ><<<(int)ROWS, 256, 0, stream>>>(T2, DEC.g1, DEC.b1n, T3); // +ln quirk
  attention(stream, XOUT, DEC.att, /*causal=*/false, Qb, Kb, Vb, Sb, Ob, T1);
  run_add(stream, XOUT, T1, T4);                              // trg + attn
  ln_ker<true ><<<(int)ROWS, 256, 0, stream>>>(T4, DEC.g1, DEC.b1n, T5); // +ln quirk
  run_add(stream, T3, T5, T6);                                // sf = src + trg
  ln_ker<false><<<(int)ROWS, 256, 0, stream>>>(T6, DEC.g2, DEC.b2n, T1);
  ffn(stream, T1, DEC, FFH, T2);
  run_add(stream, T6, T2, T6);                                // sf + ffn(ln(sf))
  ln_ker<false><<<(int)ROWS, 256, 0, stream>>>(T6, DEC.g2, DEC.b2n, T1);

  // ---- final LN + vocab projection ----------------------------------------
  ln_ker<false><<<(int)ROWS, 256, 0, stream>>>(T1, gf, bff, T2);
  rgemm<2, 2, false, true, false>(stream, T2, wf, bfw, (float*)d_out,
                                  (int)ROWS, NV, ND, ND, NV, NV);
}
